// RAN_65927747993783
// MI455X (gfx1250) — compile-verified, matrix operand order checked by reading, not run on hardware
//
#include <hip/hip_runtime.h>
#include <hip/hip_bf16.h>

// Problem constants (reference: L=2, B=32, T=128, H=E=1024, V=50257)
constexpr int L_ = 2;
constexpr int B_ = 32;
constexpr int T_ = 128;
constexpr int H_ = 1024;
constexpr int E_ = 1024;
constexpr int V_ = 50257;

typedef __attribute__((ext_vector_type(16))) __bf16 v16bf;
typedef __attribute__((ext_vector_type(8)))  __bf16 v8bf;
typedef __attribute__((ext_vector_type(8)))  float  v8f;
typedef __attribute__((ext_vector_type(4)))  int    v4i;

// ---------------------------------------------------------------------------
// Async global->LDS staging (gfx1250 GLOBAL_LOAD_ASYNC_TO_LDS_B128, ASYNCcnt).
// The builtin exists on this toolchain; its params are int4 pointers in the
// global (AS1) / LDS (AS3) address spaces. Guarded so missing-name toolchains
// still compile via the register-relay fallback.
// ---------------------------------------------------------------------------
#if defined(__has_builtin)
#if __has_builtin(__builtin_amdgcn_global_load_async_to_lds_b128)
#define HAVE_ASYNC_LDS 1
#endif
#endif
#ifndef HAVE_ASYNC_LDS
#define HAVE_ASYNC_LDS 0
#endif

typedef __attribute__((address_space(1))) v4i v4i_as1;
typedef __attribute__((address_space(3))) v4i v4i_as3;

static __device__ __forceinline__ void stage_b128(const __bf16* g, __bf16* l) {
#if HAVE_ASYNC_LDS
  __builtin_amdgcn_global_load_async_to_lds_b128((v4i_as1*)(g), (v4i_as3*)(l),
                                                 0, 0);
#else
  *(v8bf*)l = *(const v8bf*)g;
#endif
}

static __device__ __forceinline__ void stage_wait() {
#if HAVE_ASYNC_LDS
#if __has_builtin(__builtin_amdgcn_s_wait_asynccnt)
  __builtin_amdgcn_s_wait_asynccnt(0);
#else
  asm volatile("s_wait_asynccnt 0" ::: "memory");
#endif
#endif
}

// ---------------------------------------------------------------------------
// WMMA helpers (CDNA5 wave32, V_WMMA_F32_16X16X32_BF16)
// A-matrix 16x32 (MxK) per-lane layout (ISA 7.12.2):
//   lanes 0-15 : row M=lane,    elems 0..7 = K0+0..7,  elems 8..15 = K0+16..23
//   lanes16-31 : row M=lane-16, elems 0..7 = K0+8..15, elems 8..15 = K0+24..31
// B-matrix 32x16 (KxN) mirrors with N in place of M; since B = W^T, a lane's
// column n is row n of the row-major weight, contiguous along K.
// ---------------------------------------------------------------------------
static __device__ __forceinline__ v16bf
load_tile16x32(const __bf16* base, int ld, int r0, int k0, int lane) {
  int r = r0 + (lane & 15);
  int k = k0 + ((lane >> 4) << 3);
  const __bf16* p = base + (size_t)r * ld + k;
  v8bf lo = *(const v8bf*)(p);        // K .. K+7
  v8bf hi = *(const v8bf*)(p + 16);   // K+16 .. K+23
  return __builtin_shufflevector(lo, hi, 0,1,2,3,4,5,6,7,8,9,10,11,12,13,14,15);
}

static __device__ __forceinline__ v16bf
load_tile16x32_row(const __bf16* base, int ld, int row, int k0, int lane) {
  // row already includes (lane&15) and any clamping
  const __bf16* p = base + (size_t)row * ld + k0 + ((lane >> 4) << 3);
  v8bf lo = *(const v8bf*)(p);
  v8bf hi = *(const v8bf*)(p + 16);
  return __builtin_shufflevector(lo, hi, 0,1,2,3,4,5,6,7,8,9,10,11,12,13,14,15);
}

static __device__ __forceinline__ v8f wmma_bf16(v16bf a, v16bf b, v8f c) {
  return __builtin_amdgcn_wmma_f32_16x16x32_bf16(false, a, false, b, (short)0, c,
                                                 false, false);
}

// ---------------------------------------------------------------------------
// utility kernels
// ---------------------------------------------------------------------------
__global__ void f32_to_bf16_kernel(const float* __restrict__ src,
                                   __bf16* __restrict__ dst, int n) {
  for (int i = blockIdx.x * blockDim.x + threadIdx.x; i < n;
       i += gridDim.x * blockDim.x)
    dst[i] = (__bf16)src[i];
}

__global__ void zero_u32_kernel(unsigned* __restrict__ p, int n) {
  int i = blockIdx.x * blockDim.x + threadIdx.x;
  if (i < n) p[i] = 0u;
}

__global__ void embed_kernel(const int* __restrict__ word,
                             const float* __restrict__ emb,
                             __bf16* __restrict__ xa) {
  int row = blockIdx.x;              // t*B + b
  int w = word[row];
  const float* src = emb + (size_t)w * E_;
  __bf16* dst = xa + (size_t)row * E_;
  for (int e = threadIdx.x; e < E_; e += blockDim.x) dst[e] = (__bf16)src[e];
}

// ---------------------------------------------------------------------------
// Input linears: zc/zi/zf = xa @ W{c,i,f}^T + b{c,i,f}
// One wave: 2 M-tiles x 2 N-tiles x 3 gates = 12 WMMAs / K-step,
// 4 A-loads + 12 B-loads (b128) per K-step -> 1.33 loads/WMMA.
// ---------------------------------------------------------------------------
__global__ __launch_bounds__(256)
void input_linear_kernel(const __bf16* __restrict__ xa,   // [T*B, E]
                         const __bf16* __restrict__ wc,   // [H, E]
                         const __bf16* __restrict__ wi,
                         const __bf16* __restrict__ wf,
                         const float* __restrict__ bc,    // [H]
                         const float* __restrict__ bi,
                         const float* __restrict__ bff,
                         float* __restrict__ zc,          // [T*B, H]
                         float* __restrict__ zi,
                         float* __restrict__ zf) {
  int lane = threadIdx.x & 31;
  int wav  = threadIdx.x >> 5;
  int gw   = blockIdx.x * 8 + wav;    // 0..4095
  int mpair = gw >> 5;                // 0..127  (M = 4096 -> 128 pairs of 16)
  int npair = gw & 31;                // 0..31   (N = 1024 -> 32 pairs of 16)
  int m0 = mpair * 32;
  int n0 = npair * 32;

  v8f acc[3][2][2] = {};              // [gate][mt][nt]
#pragma unroll 1
  for (int k0 = 0; k0 < E_; k0 += 32) {
    v16bf a0 = load_tile16x32(xa, E_, m0,      k0, lane);
    v16bf a1 = load_tile16x32(xa, E_, m0 + 16, k0, lane);
#pragma unroll
    for (int nt = 0; nt < 2; ++nt) {
      int nb = n0 + nt * 16;
      v16bf b0 = load_tile16x32(wc, E_, nb, k0, lane);
      acc[0][0][nt] = wmma_bf16(a0, b0, acc[0][0][nt]);
      acc[0][1][nt] = wmma_bf16(a1, b0, acc[0][1][nt]);
      v16bf b1 = load_tile16x32(wi, E_, nb, k0, lane);
      acc[1][0][nt] = wmma_bf16(a0, b1, acc[1][0][nt]);
      acc[1][1][nt] = wmma_bf16(a1, b1, acc[1][1][nt]);
      v16bf b2 = load_tile16x32(wf, E_, nb, k0, lane);
      acc[2][0][nt] = wmma_bf16(a0, b2, acc[2][0][nt]);
      acc[2][1][nt] = wmma_bf16(a1, b2, acc[2][1][nt]);
    }
  }

#pragma unroll
  for (int mt = 0; mt < 2; ++mt) {
    int mrow = m0 + mt * 16 + ((lane >> 4) << 3);
#pragma unroll
    for (int nt = 0; nt < 2; ++nt) {
      int ncol = n0 + nt * 16 + (lane & 15);
      float bcv = bc[ncol], biv = bi[ncol], bfv = bff[ncol];
#pragma unroll
      for (int r = 0; r < 8; ++r) {
        size_t idx = (size_t)(mrow + r) * H_ + ncol;
        zc[idx] = acc[0][mt][nt][r] + bcv;
        zi[idx] = acc[1][mt][nt][r] + biv;
        zf[idx] = acc[2][mt][nt][r] + bfv;
      }
    }
  }
}

// ---------------------------------------------------------------------------
// Software grid barrier (monotonic counter, agent scope)
// ---------------------------------------------------------------------------
static __device__ __forceinline__ void grid_sync(unsigned* cnt, unsigned nblocks,
                                                 unsigned step) {
  __syncthreads();
  if (threadIdx.x == 0) {
    __threadfence();
    unsigned target = (step + 1u) * nblocks;
    __hip_atomic_fetch_add(cnt, 1u, __ATOMIC_ACQ_REL, __HIP_MEMORY_SCOPE_AGENT);
    while (__hip_atomic_load(cnt, __ATOMIC_ACQUIRE, __HIP_MEMORY_SCOPE_AGENT) <
           target) {
      __builtin_amdgcn_s_sleep(1);
    }
    __threadfence();
  }
  __syncthreads();
}

// ---------------------------------------------------------------------------
// Persistent recurrence kernel: one launch for all T timesteps of one layer.
//   i = sigmoid(h @ Wi^T + bi + zi[t]); f = sigmoid(h @ Wf^T + bf + zf[t])
//   l' = i*zc[t] + f*l ; h' = tanh(l')
// 64 workgroups (one per 16-col N-tile), 2 waves each (M-tiles of B=32).
// The workgroup's 16-row slices of Wi/Wf (2 x 32 KB bf16) are staged in LDS
// ONCE (via async-to-LDS) and reused across all 128 timesteps.
// ---------------------------------------------------------------------------
__global__ __launch_bounds__(64)
void ran_seq_kernel(__bf16* __restrict__ hbuf,          // [(T+1)*B, H] bf16
                    const __bf16* __restrict__ whi,     // [H,H] layer slice
                    const __bf16* __restrict__ whf,
                    const float* __restrict__ bhi,      // [H]
                    const float* __restrict__ bhf,
                    const float* __restrict__ zc,       // [T*B,H]
                    const float* __restrict__ zi,
                    const float* __restrict__ zf,
                    float* __restrict__ lstate,         // [B,H] fp32
                    float* __restrict__ fh,             // finals h [B,H]
                    float* __restrict__ fl,             // finals l [B,H]
                    unsigned* __restrict__ barrier_cnt) {
  __shared__ __attribute__((aligned(16))) __bf16 sWi[16 * H_];  // 32 KB
  __shared__ __attribute__((aligned(16))) __bf16 sWf[16 * H_];  // 32 KB

  int ntile = blockIdx.x;            // 0..63
  int n0 = ntile * 16;

  // stage this tile's weight rows into LDS (read from HBM exactly once)
  {
    const __bf16* si = whi + (size_t)n0 * H_;
    const __bf16* sf = whf + (size_t)n0 * H_;
    for (int i = threadIdx.x; i < 16 * H_ / 8; i += blockDim.x) {
      stage_b128(si + i * 8, sWi + i * 8);
      stage_b128(sf + i * 8, sWf + i * 8);
    }
    stage_wait();
  }
  __syncthreads();

  int lane  = threadIdx.x & 31;
  int mtile = threadIdx.x >> 5;      // 0..1 (B = 32)
  int m0 = mtile * 16;
  int mrow = m0 + ((lane >> 4) << 3);
  int ncol = n0 + (lane & 15);
  float bih = bhi[ncol];
  float bfh = bhf[ncol];

  for (int t = 0; t < T_; ++t) {
    const __bf16* hprev = hbuf + (size_t)t * B_ * H_;
    __bf16* hnext = hbuf + (size_t)(t + 1) * B_ * H_;

    v8f ai = {}, af = {};
#pragma unroll 1
    for (int k0 = 0; k0 < H_; k0 += 32) {
      v16bf a   = load_tile16x32(hprev, H_, m0, k0, lane);
      v16bf bi_ = load_tile16x32(sWi, H_, 0, k0, lane);
      ai = wmma_bf16(a, bi_, ai);
      v16bf bf_ = load_tile16x32(sWf, H_, 0, k0, lane);
      af = wmma_bf16(a, bf_, af);
    }

    const float* zct = zc + (size_t)t * B_ * H_;
    const float* zit = zi + (size_t)t * B_ * H_;
    const float* zft = zf + (size_t)t * B_ * H_;
#pragma unroll
    for (int r = 0; r < 8; ++r) {
      size_t idx = (size_t)(mrow + r) * H_ + ncol;
      float ig = 1.0f / (1.0f + expf(-(ai[r] + bih + zit[idx])));
      float fg = 1.0f / (1.0f + expf(-(af[r] + bfh + zft[idx])));
      float ln = ig * zct[idx] + fg * lstate[idx];
      float hn = tanhf(ln);
      lstate[idx] = ln;
      hnext[idx] = (__bf16)hn;
      if (t == T_ - 1) { fh[idx] = hn; fl[idx] = ln; }
    }

    grid_sync(barrier_cnt, gridDim.x, (unsigned)t);
  }
}

// ---------------------------------------------------------------------------
// Vocab projection: logits = x @ h2o_w^T + h2o_b ; N = 50257 (ragged tail).
// Block stages TWO 16-row A strips (64 KB of LDS) via async-to-LDS; each of
// the 8 waves computes 2 M-tiles x 4 N-tiles, so each B tile (global/L2)
// feeds 2 WMMAs -> 1 global b128 per WMMA. B (103 MB bf16) is L2-resident.
// ---------------------------------------------------------------------------
__global__ __launch_bounds__(256)
void proj_kernel(const __bf16* __restrict__ xa,    // [T*B, H]
                 const __bf16* __restrict__ w,     // [V, H] bf16
                 const float* __restrict__ bias,   // [V]
                 float* __restrict__ out) {        // [T*B, V]
  constexpr int NT = 4;
  __shared__ __attribute__((aligned(16))) __bf16 sA[32 * H_];  // 64 KB

  int mpair = blockIdx.y;                     // 0..127 (M = 4096 -> 32-row strips)
  // stage both A strips once per block
  {
    const __bf16* src = xa + (size_t)mpair * 32 * H_;
    for (int i = threadIdx.x; i < 32 * H_ / 8; i += blockDim.x)
      stage_b128(src + i * 8, sA + i * 8);
    stage_wait();
  }
  __syncthreads();

  int lane = threadIdx.x & 31;
  int wav  = threadIdx.x >> 5;
  int nbase = blockIdx.x * 32 + wav * NT;     // tile index

  int rowB[NT];
#pragma unroll
  for (int nt = 0; nt < NT; ++nt) {
    int rr = (nbase + nt) * 16 + (lane & 15);
    rowB[nt] = rr < V_ ? rr : (V_ - 1);       // clamp OOB loads
  }

  v8f acc[2][NT] = {};                        // [mt][nt]
#pragma unroll 1
  for (int k0 = 0; k0 < H_; k0 += 32) {
    v16bf a0 = load_tile16x32(sA, H_, 0,  k0, lane);   // from LDS
    v16bf a1 = load_tile16x32(sA, H_, 16, k0, lane);
#pragma unroll
    for (int nt = 0; nt < NT; ++nt) {
      const __bf16* bp = w + (size_t)rowB[nt] * H_ + k0;
      if (k0 + 32 < H_) __builtin_prefetch(bp + 32, 0, 1);
      v16bf b = load_tile16x32_row(w, H_, rowB[nt], k0, lane);
      acc[0][nt] = wmma_bf16(a0, b, acc[0][nt]);
      acc[1][nt] = wmma_bf16(a1, b, acc[1][nt]);
    }
  }

#pragma unroll
  for (int mt = 0; mt < 2; ++mt) {
    int mrow = mpair * 32 + mt * 16 + ((lane >> 4) << 3);
#pragma unroll
    for (int nt = 0; nt < NT; ++nt) {
      int ncol = (nbase + nt) * 16 + (lane & 15);
      if (ncol < V_) {
        float bv = bias[ncol];
#pragma unroll
        for (int r = 0; r < 8; ++r)
          out[(size_t)(mrow + r) * V_ + ncol] = acc[mt][nt][r] + bv;
      }
    }
  }
}

// ---------------------------------------------------------------------------
// Host launcher
// ---------------------------------------------------------------------------
extern "C" void kernel_launch(void* const* d_in, const int* in_sizes, int n_in,
                              void* d_out, int out_size, void* d_ws, size_t ws_size,
                              hipStream_t stream) {
  const int*   word  = (const int*)d_in[0];
  const float* h_s   = (const float*)d_in[1];
  const float* l_s   = (const float*)d_in[2];
  const float* emb   = (const float*)d_in[3];
  const float* h2o_w = (const float*)d_in[4];
  const float* h2o_b = (const float*)d_in[5];
  const float* w_x2c = (const float*)d_in[6];
  const float* w_x2i = (const float*)d_in[7];
  const float* w_x2f = (const float*)d_in[8];
  const float* w_h2i = (const float*)d_in[9];
  const float* w_h2f = (const float*)d_in[10];
  const float* b_x2c = (const float*)d_in[11];
  const float* b_x2i = (const float*)d_in[12];
  const float* b_x2f = (const float*)d_in[13];
  const float* b_h2i = (const float*)d_in[14];
  const float* b_h2f = (const float*)d_in[15];

  float* fh     = (float*)d_out;                       // [L,B,H]
  float* fl     = fh + (size_t)L_ * B_ * H_;           // [L,B,H]
  float* logits = fh + (size_t)2 * L_ * B_ * H_;       // [T,B,V]

  // workspace bump allocator
  char* wsp = (char*)d_ws;
  auto alloc = [&](size_t bytes) -> void* {
    void* p = wsp;
    wsp += (bytes + 255) & ~(size_t)255;
    return p;
  };
  const size_t LHE = (size_t)L_ * H_ * E_;
  const size_t LHH = (size_t)L_ * H_ * H_;
  const size_t TBH = (size_t)T_ * B_ * H_;

  __bf16* wxc_bf = (__bf16*)alloc(LHE * 2);
  __bf16* wxi_bf = (__bf16*)alloc(LHE * 2);
  __bf16* wxf_bf = (__bf16*)alloc(LHE * 2);
  __bf16* whi_bf = (__bf16*)alloc(LHH * 2);
  __bf16* whf_bf = (__bf16*)alloc(LHH * 2);
  __bf16* h2o_bf = (__bf16*)alloc((size_t)V_ * H_ * 2);
  __bf16* xa0    = (__bf16*)alloc((size_t)T_ * B_ * E_ * 2);
  __bf16* hbuf0  = (__bf16*)alloc((size_t)(T_ + 1) * B_ * H_ * 2);
  __bf16* hbuf1  = (__bf16*)alloc((size_t)(T_ + 1) * B_ * H_ * 2);
  float*  zc     = (float*)alloc(TBH * 4);
  float*  zi     = (float*)alloc(TBH * 4);
  float*  zf     = (float*)alloc(TBH * 4);
  float*  lst    = (float*)alloc((size_t)L_ * B_ * H_ * 4);
  unsigned* bars = (unsigned*)alloc(64 * 4);           // grid-barrier counters

  auto cvt = [&](const float* s, __bf16* d, size_t n) {
    int grid = (int)((n + 255) / 256);
    if (grid > 8192) grid = 8192;
    f32_to_bf16_kernel<<<grid, 256, 0, stream>>>(s, d, (int)n);
  };

  // init barrier counters (ws is NOT re-poisoned between replays)
  zero_u32_kernel<<<1, 64, 0, stream>>>(bars, 64);

  // weight / embedding conversions (amortized across the whole forward)
  cvt(w_x2c, wxc_bf, LHE);
  cvt(w_x2i, wxi_bf, LHE);
  cvt(w_x2f, wxf_bf, LHE);
  cvt(w_h2i, whi_bf, LHH);
  cvt(w_h2f, whf_bf, LHH);
  cvt(h2o_w, h2o_bf, (size_t)V_ * H_);
  embed_kernel<<<T_ * B_, 256, 0, stream>>>(word, emb, xa0);

  for (int layer = 0; layer < L_; ++layer) {
    const __bf16* xa = (layer == 0) ? xa0 : (hbuf0 + (size_t)B_ * H_);
    __bf16* hbuf = (layer == 0) ? hbuf0 : hbuf1;
    const size_t wOff = (size_t)layer * H_ * E_;
    const size_t hOff = (size_t)layer * H_ * H_;
    const size_t bOff = (size_t)layer * H_;

    input_linear_kernel<<<512, 256, 0, stream>>>(
        xa, wxc_bf + wOff, wxi_bf + wOff, wxf_bf + wOff,
        b_x2c + bOff, b_x2i + bOff, b_x2f + bOff, zc, zi, zf);

    // init recurrent state: h (bf16, slot 0) and l (fp32)
    cvt(h_s + (size_t)layer * B_ * H_, hbuf, (size_t)B_ * H_);
    (void)hipMemcpyAsync(lst + (size_t)layer * B_ * H_,
                         l_s + (size_t)layer * B_ * H_,
                         (size_t)B_ * H_ * sizeof(float),
                         hipMemcpyDeviceToDevice, stream);

    // one persistent launch for all 128 timesteps of this layer
    ran_seq_kernel<<<H_ / 16, 64, 0, stream>>>(
        hbuf, whi_bf + hOff, whf_bf + hOff, b_h2i + bOff, b_h2f + bOff,
        zc, zi, zf, lst + (size_t)layer * B_ * H_,
        fh + (size_t)layer * B_ * H_, fl + (size_t)layer * B_ * H_,
        bars + layer);
  }

  // vocab projection from layer-1 h sequence
  int ntiles = (V_ + 15) / 16;             // 3142
  int gx = (ntiles + 31) / 32;             // 99
  proj_kernel<<<dim3(gx, T_ * B_ / 32), 256, 0, stream>>>(
      hbuf1 + (size_t)B_ * H_, h2o_bf, h2o_b, logits);
}
